// Block_7911329760080
// MI455X (gfx1250) — matrix operand order repaired and checked by reading, not compiled, not run on hardware
//
#include <hip/hip_runtime.h>
#include <hip/hip_bf16.h>

// ---------------------------------------------------------------------------
// Transformer block for MI455X (gfx1250, wave32, WMMA + Tensor Data Mover).
// GEMMs: f16 in / f32 accum via v_wmma_f32_16x16x32_f16, tiles DMA'd into LDS
// by TENSOR_LOAD_TO_LDS (double-buffered, TENSORcnt-pipelined).
// Attention: flash-style streaming softmax, WMMA for QK^T and PV.
// ---------------------------------------------------------------------------

typedef __attribute__((ext_vector_type(16))) _Float16 v16h;
typedef __attribute__((ext_vector_type(8)))  _Float16 v8h;
typedef __attribute__((ext_vector_type(8)))  float    v8f;
typedef unsigned int u32x4 __attribute__((ext_vector_type(4)));
typedef int          i32x8 __attribute__((ext_vector_type(8)));
typedef int          i32x4 __attribute__((ext_vector_type(4)));

#define N_EMBD 1024
#define N_HEAD 16
#define HD     64
#define BSZ    2
#define TLEN   2048
#define M_TOK  (BSZ * TLEN)   // 4096 token rows

// ---- helpers --------------------------------------------------------------

__device__ inline v16h ld_frag(const _Float16* p0, const _Float16* p1) {
  union { v16h v; v8h h[2]; } u;
  u.h[0] = *(const v8h*)p0;
  u.h[1] = *(const v8h*)p1;
  return u.v;
}

__device__ inline v8f wmma_f16(v16h a, v16h b, v8f c) {
  // (neg_a, A, neg_b, B, c_mod, C, reuse_a, reuse_b)
  return __builtin_amdgcn_wmma_f32_16x16x32_f16(false, a, false, b, (short)0, c,
                                                false, false);
}

__device__ inline float redmax16(float x) {
  x = fmaxf(x, __shfl_xor(x, 1, 32));
  x = fmaxf(x, __shfl_xor(x, 2, 32));
  x = fmaxf(x, __shfl_xor(x, 4, 32));
  x = fmaxf(x, __shfl_xor(x, 8, 32));
  return x;
}
__device__ inline float redsum16(float x) {
  x += __shfl_xor(x, 1, 32);
  x += __shfl_xor(x, 2, 32);
  x += __shfl_xor(x, 4, 32);
  x += __shfl_xor(x, 8, 32);
  return x;
}

__device__ inline unsigned lds_offset_of(const void* p) {
  // Generic pointer to LDS: low 32 bits are the LDS byte offset (ISA §10.2).
  return (unsigned)(unsigned long long)(uintptr_t)p;
}

// Issue one TDM 2D tile load: tile_dim0=32 halves (one K-step row),
// tile_dim1=128 rows, row stride K halves in memory, LDS rows padded by the
// DMA itself to a 40-half stride (pad_interval=16 DWORDs, pad_amount=4 DWORDs).
__device__ inline void tdm_issue_tile_32x128(const _Float16* gptr,
                                             unsigned lds_byte_off,
                                             int k_stride_elems) {
  unsigned long long ga = (unsigned long long)(uintptr_t)gptr;
  u32x4 g0;
  g0[0] = 1u;                                   // count=1 (valid), user mode
  g0[1] = lds_byte_off;                         // lds_addr
  g0[2] = (unsigned)(ga & 0xFFFFFFFFu);         // global_addr[31:0]
  g0[3] = (unsigned)((ga >> 32) & 0x01FFFFFFu)  // global_addr[56:32]
        | 0x80000000u;                          // type=2 ("image")
  i32x8 g1;
  g1[0] = (1 << 16)        // data_size = 2 bytes
        | (1 << 20)        // pad_enable
        | (3 << 22)        // pad_interval: 16 DWORDs (32 halves) ...
        | (3 << 25);       // ... then pad 4 DWORDs (8 halves) -> stride 40
  g1[1] = (int)0xFFFF0000; // tensor_dim0 = 0x7FFFFFFF (low 16 here; never OOB)
  g1[2] = (int)0xFFFF7FFF; // tensor_dim0 hi16 | tensor_dim1 low16
  g1[3] = (int)(0x7FFF | (32 << 16));   // tensor_dim1 hi16 | tile_dim0=32
  g1[4] = 128;                          // tile_dim1=128, tile_dim2=0
  g1[5] = k_stride_elems;               // tensor_dim0_stride (elements)
  g1[6] = 0;
  g1[7] = 0;
  i32x4 z4 = {0, 0, 0, 0};
  i32x8 z8 = {0, 0, 0, 0, 0, 0, 0, 0};
  // clang-23 / therock-10.0 form: (g0, g1, g2, g3, extra, cpol)
  __builtin_amdgcn_tensor_load_to_lds(g0, g1, z4, z4, z8, 0);
}

// ---- weight convert + transpose: fp32 W[K][N] -> f16 Wt[N][K] -------------

__global__ void convert_transpose_kernel(const float* __restrict__ W,
                                         _Float16* __restrict__ Wt,
                                         int K, int N) {
  long idx = (long)blockIdx.x * blockDim.x + threadIdx.x;
  long total = (long)K * N;
  if (idx >= total) return;
  int n = (int)(idx / K);
  int k = (int)(idx % K);
  Wt[idx] = (_Float16)W[(long)k * N + n];
}

// ---- LayerNorm: fp32 in -> f16 out (one row per block) --------------------

__global__ void __launch_bounds__(256)
layernorm_f16_kernel(const float* __restrict__ x, const float* __restrict__ g,
                     const float* __restrict__ b, _Float16* __restrict__ out) {
  __shared__ float red[256];
  const int row = blockIdx.x, tid = threadIdx.x;
  const float* xr = x + (long)row * N_EMBD;
  float v[4];
  float s = 0.f;
#pragma unroll
  for (int i = 0; i < 4; ++i) { v[i] = xr[tid + i * 256]; s += v[i]; }
  red[tid] = s; __syncthreads();
  for (int off = 128; off > 0; off >>= 1) {
    if (tid < off) red[tid] += red[tid + off];
    __syncthreads();
  }
  const float mean = red[0] * (1.f / N_EMBD);
  __syncthreads();
  float ss = 0.f;
#pragma unroll
  for (int i = 0; i < 4; ++i) { float d = v[i] - mean; ss += d * d; }
  red[tid] = ss; __syncthreads();
  for (int off = 128; off > 0; off >>= 1) {
    if (tid < off) red[tid] += red[tid + off];
    __syncthreads();
  }
  const float rstd = rsqrtf(red[0] * (1.f / N_EMBD) + 1e-5f);
#pragma unroll
  for (int i = 0; i < 4; ++i) {
    int c = tid + i * 256;
    out[(long)row * N_EMBD + c] = (_Float16)((v[i] - mean) * rstd * g[c] + b[c]);
  }
}

// ---- WMMA GEMM with TDM double buffering ----------------------------------
// C[M,N] = epilogue(A[M,K] @ B[K,N] + bias)
// A: f16 row-major. Bt: f16 B transposed [N][K] (K contiguous).
// Block tile 128x128, 8 waves each computing 32x64 (2x4 WMMA tiles), K-step 32.
// Wave 0 drives the Tensor Data Mover: tile k+1 streams into the spare buffer
// while all waves run WMMA on tile k; completion via s_wait_tensorcnt.
// EPI 0: f16 out, +bias.  EPI 1: f16 out, gelu(x+bias).  EPI 2: f32 out, resid+bias.

template <int EPI>
__global__ void __launch_bounds__(256)
gemm_wmma_kernel(const _Float16* __restrict__ A, const _Float16* __restrict__ Bt,
                 const float* __restrict__ bias, const float* __restrict__ resid,
                 void* __restrict__ Cout, int M, int N, int K) {
  __shared__ _Float16 ldsA[2][128 * 40];   // stride 40 halves (pad via TDM)
  __shared__ _Float16 ldsB[2][128 * 40];

  const int tid  = threadIdx.x;
  const int lane = tid & 31;
  const int wave = tid >> 5;
  const int wr   = wave >> 1;          // wave row   0..3 -> m offset wr*32
  const int wc   = wave & 1;           // wave col   0..1 -> n offset wc*64
  const int hlf  = lane >> 4;          // lane half (0/1)
  const int l16  = lane & 15;
  const int m0   = blockIdx.y * 128;
  const int n0   = blockIdx.x * 128;

  const v8f zero = {0.f, 0.f, 0.f, 0.f, 0.f, 0.f, 0.f, 0.f};
  v8f acc[2][4];
#pragma unroll
  for (int i = 0; i < 2; ++i)
#pragma unroll
    for (int j = 0; j < 4; ++j) acc[i][j] = zero;

  const int nk = K >> 5;   // number of 32-wide K-steps

  // prologue: DMA tile 0 into buffer 0
  if (wave == 0) {
    tdm_issue_tile_32x128(A  + (long)m0 * K, lds_offset_of(&ldsA[0][0]), K);
    tdm_issue_tile_32x128(Bt + (long)n0 * K, lds_offset_of(&ldsB[0][0]), K);
  }

  for (int it = 0; it < nk; ++it) {
    const int c = it & 1;
    if (wave == 0) {
      if (it + 1 < nk) {
        // stream next tile into the spare buffer, then wait for current tile
        const int kk = (it + 1) << 5;
        tdm_issue_tile_32x128(A  + (long)m0 * K + kk,
                              lds_offset_of(&ldsA[c ^ 1][0]), K);
        tdm_issue_tile_32x128(Bt + (long)n0 * K + kk,
                              lds_offset_of(&ldsB[c ^ 1][0]), K);
        __builtin_amdgcn_s_wait_tensorcnt(2);
      } else {
        __builtin_amdgcn_s_wait_tensorcnt(0);
      }
    }
    __syncthreads();   // current buffer visible to all waves

    const _Float16* La = &ldsA[c][0];
    const _Float16* Lb = &ldsB[c][0];

    // A fragments: lanes 0-15 hold K 0-7 & 16-23, lanes 16-31 hold K 8-15 & 24-31
    v16h af[2];
#pragma unroll
    for (int mt = 0; mt < 2; ++mt) {
      int r = wr * 32 + mt * 16 + l16;
      af[mt] = ld_frag(&La[r * 40 + hlf * 8], &La[r * 40 + 16 + hlf * 8]);
    }
    // B fragments: lane column fixed, same K split as A (K 0-7 & 16-23 / 8-15 & 24-31)
#pragma unroll
    for (int nt = 0; nt < 4; ++nt) {
      int cc = wc * 64 + nt * 16 + l16;
      v16h bf = ld_frag(&Lb[cc * 40 + hlf * 8], &Lb[cc * 40 + 16 + hlf * 8]);
      acc[0][nt] = wmma_f16(af[0], bf, acc[0][nt]);
      acc[1][nt] = wmma_f16(af[1], bf, acc[1][nt]);
    }
    __syncthreads();   // all waves done with buffer c; TDM may overwrite next iter
  }

  // epilogue: C/D layout -> lane l16 = column, VGPR v + hlf*8 = row
#pragma unroll
  for (int nt = 0; nt < 4; ++nt) {
    const int n = n0 + wc * 64 + nt * 16 + l16;
    const float bn = bias[n];
#pragma unroll
    for (int mt = 0; mt < 2; ++mt) {
#pragma unroll
      for (int v = 0; v < 8; ++v) {
        const int m = m0 + wr * 32 + mt * 16 + v + hlf * 8;
        const long idx = (long)m * N + n;
        float val = acc[mt][nt][v] + bn;
        if constexpr (EPI == 0) {
          ((_Float16*)Cout)[idx] = (_Float16)val;
        } else if constexpr (EPI == 1) {
          float gel = 0.5f * val * (1.f + erff(val * 0.70710678118654752f));
          ((_Float16*)Cout)[idx] = (_Float16)gel;
        } else {
          ((float*)Cout)[idx] = resid[idx] + val;
        }
      }
    }
  }
}

// ---- Flash attention (full, non-causal per reference) ---------------------
// Grid: (T/64 q-tiles, heads, batch). Block: 128 threads (4 waves).
// Each wave owns 16 q rows; key tiles of 64 streamed through LDS.

__global__ void __launch_bounds__(128)
attention_kernel(const _Float16* __restrict__ q, const _Float16* __restrict__ k,
                 const _Float16* __restrict__ v, _Float16* __restrict__ y) {
  __shared__ _Float16 ldsK[64 * 72];       // [key][d]
  __shared__ _Float16 ldsV[64 * 72];       // transposed: [d][key]
  __shared__ _Float16 ldsP[4 * 16 * 72];   // per-wave P staging [row][key]

  const int tid  = threadIdx.x;
  const int lane = tid & 31;
  const int wave = tid >> 5;
  const int hlf  = lane >> 4;
  const int l16  = lane & 15;
  const int qt   = blockIdx.x;
  const int h    = blockIdx.y;
  const int b    = blockIdx.z;
  const long rowbase = (long)b * TLEN;
  const int  hoff    = h * HD;
  const int  qrow0   = qt * 64 + wave * 16;

  // Q fragments (held in registers for the whole kernel): hd=64 -> 2 K-steps
  v16h qf[2];
#pragma unroll
  for (int s = 0; s < 2; ++s) {
    const _Float16* p =
        &q[(rowbase + qrow0 + l16) * N_EMBD + hoff + s * 32 + hlf * 8];
    qf[s] = ld_frag(p, p + 16);
  }

  const v8f zero = {0.f, 0.f, 0.f, 0.f, 0.f, 0.f, 0.f, 0.f};
  v8f oacc[4];
  float mrow[8], lrow[8];
#pragma unroll
  for (int j = 0; j < 4; ++j) oacc[j] = zero;
#pragma unroll
  for (int vv = 0; vv < 8; ++vv) { mrow[vv] = -1.0e30f; lrow[vv] = 0.f; }

  _Float16* myP = &ldsP[wave * 16 * 72];
  const float scale = 0.125f;   // 1/sqrt(64)

  for (int kt = 0; kt < TLEN / 64; ++kt) {
    // stage K tile [key][d] (direct copy)
    {
      const int key = tid >> 1, part = (tid & 1) * 32;
      const _Float16* src = &k[(rowbase + kt * 64 + key) * N_EMBD + hoff + part];
#pragma unroll
      for (int j = 0; j < 4; ++j)
        *(uint4*)&ldsK[key * 72 + part + j * 8] = *(const uint4*)&src[j * 8];
    }
    // stage V tile transposed [d][key]: pack key pairs into b32 stores
    {
      const int p2 = tid & 31;   // key pair 0..31 -> keys 2p2, 2p2+1
      const int g  = tid >> 5;   // d group 0..3  -> d = g*16 .. g*16+15
      union U { uint4 u[2]; unsigned short s[16]; } r0, r1;
      const _Float16* s0 = &v[(rowbase + kt * 64 + 2 * p2) * N_EMBD + hoff + g * 16];
      const _Float16* s1 = s0 + N_EMBD;
      r0.u[0] = *(const uint4*)s0;      r0.u[1] = *(const uint4*)(s0 + 8);
      r1.u[0] = *(const uint4*)s1;      r1.u[1] = *(const uint4*)(s1 + 8);
#pragma unroll
      for (int j = 0; j < 16; ++j) {
        unsigned int packed = (unsigned int)r0.s[j] | ((unsigned int)r1.s[j] << 16);
        *(unsigned int*)&ldsV[(g * 16 + j) * 72 + 2 * p2] = packed;
      }
    }
    __syncthreads();

    // S = Q @ K^T  (16 x 64 per wave; N-dim = keys, K-dim = hd)
    v8f sacc[4] = {zero, zero, zero, zero};
#pragma unroll
    for (int s = 0; s < 2; ++s) {
#pragma unroll
      for (int j = 0; j < 4; ++j) {
        const int ck = j * 16 + l16;
        v16h bf = ld_frag(&ldsK[ck * 72 + s * 32 + hlf * 8],
                          &ldsK[ck * 72 + s * 32 + hlf * 8 + 16]);
        sacc[j] = wmma_f16(qf[s], bf, sacc[j]);
      }
    }

    // online softmax (row = VGPR index + hlf*8; columns live across 16 lanes)
#pragma unroll
    for (int vv = 0; vv < 8; ++vv) {
      float tmax = -1.0e30f;
#pragma unroll
      for (int j = 0; j < 4; ++j) tmax = fmaxf(tmax, sacc[j][vv]);
      tmax = redmax16(tmax) * scale;
      const float mnew = fmaxf(mrow[vv], tmax);
      const float corr = __expf(mrow[vv] - mnew);
      mrow[vv] = mnew;
      const int prow = vv + hlf * 8;
      float rsum = 0.f;
#pragma unroll
      for (int j = 0; j < 4; ++j) {
        float p = __expf(sacc[j][vv] * scale - mnew);
        rsum += p;
        myP[prow * 72 + j * 16 + l16] = (_Float16)p;
      }
      rsum = redsum16(rsum);
      lrow[vv] = lrow[vv] * corr + rsum;
#pragma unroll
      for (int j = 0; j < 4; ++j) oacc[j][vv] *= corr;
    }

    // O += P @ V   (K-dim = 64 keys -> 2 K-steps; N-dim = hd)
#pragma unroll
    for (int s = 0; s < 2; ++s) {
      const _Float16* pp = &myP[l16 * 72 + s * 32 + hlf * 8];
      v16h pf = ld_frag(pp, pp + 16);
#pragma unroll
      for (int j = 0; j < 4; ++j) {
        const int cd = j * 16 + l16;
        v16h bf = ld_frag(&ldsV[cd * 72 + s * 32 + hlf * 8],
                          &ldsV[cd * 72 + s * 32 + hlf * 8 + 16]);
        oacc[j] = wmma_f16(pf, bf, oacc[j]);
      }
    }
    __syncthreads();
  }

  // normalize and write y (f16)
#pragma unroll
  for (int vv = 0; vv < 8; ++vv) {
    const float inv = 1.f / lrow[vv];
    const int m = qrow0 + vv + hlf * 8;
#pragma unroll
    for (int j = 0; j < 4; ++j) {
      y[(rowbase + m) * N_EMBD + hoff + j * 16 + l16] =
          (_Float16)(oacc[j][vv] * inv);
    }
  }
}

// ---------------------------------------------------------------------------

extern "C" void kernel_launch(void* const* d_in, const int* in_sizes, int n_in,
                              void* d_out, int out_size, void* d_ws, size_t ws_size,
                              hipStream_t stream) {
  const float* x    = (const float*)d_in[0];
  const float* Wq   = (const float*)d_in[1];
  const float* bq   = (const float*)d_in[2];
  const float* Wk   = (const float*)d_in[3];
  const float* bk   = (const float*)d_in[4];
  const float* Wv   = (const float*)d_in[5];
  const float* bv   = (const float*)d_in[6];
  const float* Wo   = (const float*)d_in[7];
  const float* bo   = (const float*)d_in[8];
  const float* W1   = (const float*)d_in[9];
  const float* b1   = (const float*)d_in[10];
  const float* W2   = (const float*)d_in[11];
  const float* b2   = (const float*)d_in[12];
  const float* ln1g = (const float*)d_in[13];
  const float* ln1b = (const float*)d_in[14];
  const float* ln2g = (const float*)d_in[15];
  const float* ln2b = (const float*)d_in[16];

  char* ws = (char*)d_ws;
  size_t off = 0;
  auto alloc = [&](size_t bytes) -> void* {
    void* p = ws + off;
    off += (bytes + 255) & ~(size_t)255;
    return p;
  };

  _Float16* Wqt = (_Float16*)alloc((size_t)N_EMBD * N_EMBD * 2);
  _Float16* Wkt = (_Float16*)alloc((size_t)N_EMBD * N_EMBD * 2);
  _Float16* Wvt = (_Float16*)alloc((size_t)N_EMBD * N_EMBD * 2);
  _Float16* Wot = (_Float16*)alloc((size_t)N_EMBD * N_EMBD * 2);
  _Float16* W1t = (_Float16*)alloc((size_t)N_EMBD * 4 * N_EMBD * 2);
  _Float16* W2t = (_Float16*)alloc((size_t)N_EMBD * 4 * N_EMBD * 2);
  _Float16* xn1 = (_Float16*)alloc((size_t)M_TOK * N_EMBD * 2);
  _Float16* qb  = (_Float16*)alloc((size_t)M_TOK * N_EMBD * 2);
  _Float16* kb  = (_Float16*)alloc((size_t)M_TOK * N_EMBD * 2);
  _Float16* vb  = (_Float16*)alloc((size_t)M_TOK * N_EMBD * 2);
  _Float16* yb  = (_Float16*)alloc((size_t)M_TOK * N_EMBD * 2);
  float*    x2  = (float*)   alloc((size_t)M_TOK * N_EMBD * 4);
  _Float16* xn2 = (_Float16*)alloc((size_t)M_TOK * N_EMBD * 2);
  _Float16* h1  = (_Float16*)alloc((size_t)M_TOK * 4 * N_EMBD * 2);

  // weight convert + transpose (one-time per call)
  {
    long e1 = (long)N_EMBD * N_EMBD;
    long e4 = (long)N_EMBD * 4 * N_EMBD;
    convert_transpose_kernel<<<(e1 + 255) / 256, 256, 0, stream>>>(Wq, Wqt, N_EMBD, N_EMBD);
    convert_transpose_kernel<<<(e1 + 255) / 256, 256, 0, stream>>>(Wk, Wkt, N_EMBD, N_EMBD);
    convert_transpose_kernel<<<(e1 + 255) / 256, 256, 0, stream>>>(Wv, Wvt, N_EMBD, N_EMBD);
    convert_transpose_kernel<<<(e1 + 255) / 256, 256, 0, stream>>>(Wo, Wot, N_EMBD, N_EMBD);
    convert_transpose_kernel<<<(e4 + 255) / 256, 256, 0, stream>>>(W1, W1t, N_EMBD, 4 * N_EMBD);
    convert_transpose_kernel<<<(e4 + 255) / 256, 256, 0, stream>>>(W2, W2t, 4 * N_EMBD, N_EMBD);
  }

  // LN1 -> f16
  layernorm_f16_kernel<<<M_TOK, 256, 0, stream>>>(x, ln1g, ln1b, xn1);

  // QKV projections
  dim3 gP(N_EMBD / 128, M_TOK / 128);   // (8, 32)
  gemm_wmma_kernel<0><<<gP, 256, 0, stream>>>(xn1, Wqt, bq, nullptr, qb, M_TOK, N_EMBD, N_EMBD);
  gemm_wmma_kernel<0><<<gP, 256, 0, stream>>>(xn1, Wkt, bk, nullptr, kb, M_TOK, N_EMBD, N_EMBD);
  gemm_wmma_kernel<0><<<gP, 256, 0, stream>>>(xn1, Wvt, bv, nullptr, vb, M_TOK, N_EMBD, N_EMBD);

  // attention (full/non-causal per reference)
  attention_kernel<<<dim3(TLEN / 64, N_HEAD, BSZ), 128, 0, stream>>>(qb, kb, vb, yb);

  // x2 = x + y @ Wo + bo
  gemm_wmma_kernel<2><<<gP, 256, 0, stream>>>(yb, Wot, bo, x, x2, M_TOK, N_EMBD, N_EMBD);

  // LN2 -> f16
  layernorm_f16_kernel<<<M_TOK, 256, 0, stream>>>(x2, ln2g, ln2b, xn2);

  // h1 = gelu(xn2 @ W1 + b1)
  dim3 gF(4 * N_EMBD / 128, M_TOK / 128);  // (32, 32)
  gemm_wmma_kernel<1><<<gF, 256, 0, stream>>>(xn2, W1t, b1, nullptr, h1, M_TOK, 4 * N_EMBD, N_EMBD);

  // out = x2 + h1 @ W2 + b2  (fp32)
  gemm_wmma_kernel<2><<<gP, 256, 0, stream>>>(h1, W2t, b2, x2, (float*)d_out, M_TOK, N_EMBD, 4 * N_EMBD);
}
